// CausalSelfAttention_24352464570121
// MI455X (gfx1250) — matrix operand order repaired and checked by reading, not compiled, not run on hardware
//
#include <hip/hip_runtime.h>
#include <hip/hip_bf16.h>

#define B_  4
#define S_  2048
#define D_  1024
#define H_  16
#define HD_ 64

typedef __attribute__((ext_vector_type(16))) __bf16 v16bf;
typedef __attribute__((ext_vector_type(8)))  __bf16 v8bf;
typedef __attribute__((ext_vector_type(4)))  __bf16 v4bf;
typedef __attribute__((ext_vector_type(8)))  float  v8f;

#define WMMA_BF16(a, b, c) \
  __builtin_amdgcn_wmma_f32_16x16x32_bf16(false, (a), false, (b), (short)0, (c), false, false)

// f32 -> bf16 round-to-nearest-even
__device__ __forceinline__ __bf16 f2bf(float f) {
  unsigned u = __builtin_bit_cast(unsigned, f);
  u += 0x7FFFu + ((u >> 16) & 1u);
  unsigned short s = (unsigned short)(u >> 16);
  return __builtin_bit_cast(__bf16, s);
}

// A-fragment loader: two contiguous 16B segments (global or LDS).
__device__ __forceinline__ v16bf load_frag2(const __bf16* p0, const __bf16* p1) {
  v8bf lo = *(const v8bf*)p0;
  v8bf hi = *(const v8bf*)p1;
  v16bf r;
#pragma unroll
  for (int i = 0; i < 8; ++i) { r[i] = lo[i]; r[i + 8] = hi[i]; }
  return r;
}

// ---------------------------------------------------------------------------
// Cooperative async staging of a 16-row x D_ bf16 tile (32 KB) into LDS via
// the CDNA5 async DMA path: global_load_async_to_lds_b128 (ASYNCcnt-tracked),
// then s_wait_asynccnt + block barrier.  256 threads x 8 x 16B = 32 KB.
// ---------------------------------------------------------------------------
__device__ __forceinline__ void stage16_async(const __bf16* __restrict__ gsrc,
                                              __bf16* lds_tile, int tid) {
  const unsigned lbase = (unsigned)(unsigned long long)(uintptr_t)lds_tile;
#pragma unroll
  for (int i = 0; i < 8; ++i) {
    const int cch = tid + i * 256;          // 0..2047 16B chunks
    const int r   = cch >> 7;               // row 0..15
    const int kk  = (cch & 127) * 8;        // element column
    const unsigned la = lbase + (unsigned)((r * D_ + kk) * 2);
    const unsigned long long ga =
        (unsigned long long)(uintptr_t)(gsrc + (size_t)r * D_ + kk);
    asm volatile("global_load_async_to_lds_b128 %0, %1, off"
                 :: "v"(la), "v"(ga) : "memory");
  }
  asm volatile("s_wait_asynccnt 0x0" ::: "memory");
  __syncthreads();
}

__global__ __launch_bounds__(256) void cvt_bf16x4_kernel(const float* __restrict__ in,
                                                         __bf16* __restrict__ out, int n4) {
  int i = blockIdx.x * blockDim.x + threadIdx.x;
  if (i < n4) {
    float4 f = ((const float4*)in)[i];
    v4bf o;
    o[0] = f2bf(f.x); o[1] = f2bf(f.y); o[2] = f2bf(f.z); o[3] = f2bf(f.w);
    *(v4bf*)(out + 4 * (size_t)i) = o;
  }
}

// ---------------------------------------------------------------------------
// Q/K/V projection (y = x @ W^T) + RoPE.  One wave per (b, 16-row s-tile,
// head).  All 8 waves of a block share (b, stile) -> x tile staged in LDS via
// async DMA; weight fragments ping-pong between two register buffers so loads
// for the next weight overlap WMMAs of the current one.
// Outputs: Q,K (B,H,S,Dh) bf16 ; V transposed (B,H,Dh,S) bf16.
// ---------------------------------------------------------------------------
__global__ __launch_bounds__(256) void qkv_rope_kernel(
    const __bf16* __restrict__ xb,
    const __bf16* __restrict__ wq,
    const __bf16* __restrict__ wk,
    const __bf16* __restrict__ wv,
    __bf16* __restrict__ qo,
    __bf16* __restrict__ ko,
    __bf16* __restrict__ vto) {
  __shared__ __align__(16) __bf16 xtile[16 * D_];   // 32 KB

  const int lane  = threadIdx.x & 31;
  const int gw    = (blockIdx.x * blockDim.x + threadIdx.x) >> 5;
  const int head  = gw & 15;
  const int stile = (gw >> 4) & 127;
  const int b     = gw >> 11;

  const int arow = lane & 15;
  const int hi   = lane >> 4;
  const int koff = hi * 8;
  const int ncol = lane & 15;
  const int kgrp = hi * 16;

  stage16_async(xb + (size_t)(b * S_ + stile * 16) * D_, xtile, threadIdx.x);
  const __bf16* xrow_l = xtile + arow * D_;

  v8f aq[4] = {}, ak[4] = {}, av[4] = {};
  v16bf fA[4], fB[4];
  v16bf a = load_frag2(xrow_l + kgrp, xrow_l + kgrp + 8);

  for (int k = 0; k < D_; k += 32) {
    const size_t wro0 = (size_t)(head * HD_ + ncol) * D_ + k + kgrp;
#pragma unroll
    for (int g = 0; g < 4; ++g)            // Wq fragments
      fA[g] = *(const v16bf*)(wq + wro0 + (size_t)(g * 16) * D_);
#pragma unroll
    for (int g = 0; g < 4; ++g)            // Wk loads issue under Wq WMMAs
      fB[g] = *(const v16bf*)(wk + wro0 + (size_t)(g * 16) * D_);
#pragma unroll
    for (int g = 0; g < 4; ++g) aq[g] = WMMA_BF16(a, fA[g], aq[g]);
#pragma unroll
    for (int g = 0; g < 4; ++g)            // Wv loads issue under Wk WMMAs
      fA[g] = *(const v16bf*)(wv + wro0 + (size_t)(g * 16) * D_);
#pragma unroll
    for (int g = 0; g < 4; ++g) ak[g] = WMMA_BF16(a, fB[g], ak[g]);
    // next A fragment from LDS (cheap) ahead of the Wv WMMAs
    v16bf an = a;
    if (k + 32 < D_)
      an = load_frag2(xrow_l + k + 32 + kgrp, xrow_l + k + 40 + kgrp);
#pragma unroll
    for (int g = 0; g < 4; ++g) av[g] = WMMA_BF16(a, fA[g], av[g]);
    a = an;
  }

  // RoPE: pair (d, d+32) == accumulator (g, g+2), same lane slot.
  const size_t qkbase = ((size_t)(b * H_ + head) * S_ + stile * 16) * HD_;
  const size_t vtbase = (size_t)(b * H_ + head) * HD_ * S_;
#pragma unroll
  for (int r = 0; r < 8; ++r) {
    const int   srow = r + 8 * hi;
    const float t    = (float)(stile * 16 + srow);
#pragma unroll
    for (int g = 0; g < 2; ++g) {
      const int   j   = g * 16 + ncol;                                  // 0..31
      const float ang = t * __expf(-(float)j * (9.2103403720f / 32.0f)); // 10000^(-2j/64)
      const float c = __cosf(ang), s = __sinf(ang);
      float q1 = aq[g][r], q2 = aq[g + 2][r];
      aq[g][r]     = q1 * c - q2 * s;
      aq[g + 2][r] = q2 * c + q1 * s;
      float k1 = ak[g][r], k2 = ak[g + 2][r];
      ak[g][r]     = k1 * c - k2 * s;
      ak[g + 2][r] = k2 * c + k1 * s;
    }
#pragma unroll
    for (int g = 0; g < 4; ++g) {
      const int d = g * 16 + ncol;
      qo[qkbase + (size_t)srow * HD_ + d] = f2bf(aq[g][r]);
      ko[qkbase + (size_t)srow * HD_ + d] = f2bf(ak[g][r]);
      vto[vtbase + (size_t)d * S_ + stile * 16 + srow] = f2bf(av[g][r]);
    }
  }
}

// ---------------------------------------------------------------------------
// Flash attention: one wave per (b, head, 16-row query tile); online softmax
// over 32-wide key chunks bounded by the causal limit.  V fragments load
// before the softmax VALU work so exp/shuffle latency hides them; next
// chunk's K rows are prefetched into cache.
// ---------------------------------------------------------------------------
__global__ __launch_bounds__(256) void attn_kernel(
    const __bf16* __restrict__ q,
    const __bf16* __restrict__ k,
    const __bf16* __restrict__ vt,
    __bf16* __restrict__ ao) {
  __shared__ __align__(32) __bf16 plds[8][16 * 32];   // per-wave P transpose buffer

  const int lane  = threadIdx.x & 31;
  const int wid   = threadIdx.x >> 5;
  const int gw    = blockIdx.x * 8 + wid;
  const int stile = gw & 127;
  const int head  = (gw >> 7) & 15;
  const int b     = gw >> 11;
  const int s_base = stile * 16;

  const size_t bh = (size_t)(b * H_ + head);
  const __bf16* qb = q  + bh * S_ * HD_;
  const __bf16* kb = k  + bh * S_ * HD_;
  const __bf16* vb = vt + bh * HD_ * S_;

  const int arow = lane & 15;
  const int hi   = lane >> 4;
  const int koff = hi * 8;
  const int ncol = lane & 15;
  const int kgrp = hi * 16;

  const __bf16* qrow = qb + (size_t)(s_base + arow) * HD_;
  const v16bf qa0 = load_frag2(qrow + kgrp,      qrow + 8 + kgrp);
  const v16bf qa1 = load_frag2(qrow + 32 + kgrp, qrow + 40 + kgrp);

  v8f o[4] = {};
  float m[8], l[8];
#pragma unroll
  for (int r = 0; r < 8; ++r) { m[r] = -__builtin_inff(); l[r] = 0.0f; }

  const int   nchunk = (s_base + 15) / 32 + 1;
  const float scale  = 0.125f;   // 1/sqrt(64)

  for (int c = 0; c < nchunk; ++c) {
    const int tbase = c * 32;
    // K^T fragments: b[h] = K[t][kstep+kgrp+h] -> contiguous 32B each.
    const __bf16* kr0 = kb + (size_t)(tbase + ncol) * HD_ + kgrp;
    const __bf16* kr1 = kb + (size_t)(tbase + 16 + ncol) * HD_ + kgrp;
    v16bf kf0 = *(const v16bf*)(kr0);
    v16bf kf1 = *(const v16bf*)(kr0 + 32);
    v16bf kf2 = *(const v16bf*)(kr1);
    v16bf kf3 = *(const v16bf*)(kr1 + 32);
    // V fragments (independent of scores): issue now, consume after softmax.
    v16bf vf[4];
#pragma unroll
    for (int g = 0; g < 4; ++g)
      vf[g] = *(const v16bf*)(vb + (size_t)(g * 16 + ncol) * S_ + tbase + kgrp);
    if (c + 1 < nchunk) {                   // warm cache for next chunk's keys
      __builtin_prefetch(kr0 + 32 * HD_, 0, 3);
      __builtin_prefetch(kr1 + 32 * HD_, 0, 3);
    }

    v8f s0 = {}, s1 = {};
    s0 = WMMA_BF16(qa0, kf0, s0);
    s0 = WMMA_BF16(qa1, kf1, s0);
    s1 = WMMA_BF16(qa0, kf2, s1);
    s1 = WMMA_BF16(qa1, kf3, s1);

#pragma unroll
    for (int r = 0; r < 8; ++r) {
      const int srow = s_base + r + 8 * hi;
      const int t0 = tbase + ncol;
      float v0 = (t0      <= srow) ? s0[r] * scale : -__builtin_inff();
      float v1 = (t0 + 16 <= srow) ? s1[r] * scale : -__builtin_inff();
      float mx = fmaxf(v0, v1);
#pragma unroll
      for (int off = 1; off < 16; off <<= 1)      // row lives on one 16-lane group
        mx = fmaxf(mx, __shfl_xor(mx, off, 32));
      const float mn   = fmaxf(m[r], mx);
      const float corr = __expf(m[r] - mn);
      const float p0 = __expf(v0 - mn);
      const float p1 = __expf(v1 - mn);
      float ps = p0 + p1;
#pragma unroll
      for (int off = 1; off < 16; off <<= 1)
        ps += __shfl_xor(ps, off, 32);
      l[r] = l[r] * corr + ps;
      m[r] = mn;
#pragma unroll
      for (int g = 0; g < 4; ++g) o[g][r] *= corr;
      const int prow = r + 8 * hi;               // C-layout -> LDS (row, t)
      plds[wid][prow * 32 + ncol]      = f2bf(p0);
      plds[wid][prow * 32 + 16 + ncol] = f2bf(p1);
    }
    // cross-lane LDS hand-off within the wave (stores -> A-frag loads)
    asm volatile("s_wait_dscnt 0x0" ::: "memory");
    const __bf16* pl = &plds[wid][arow * 32];
    v16bf pa = load_frag2(pl + kgrp, pl + 8 + kgrp);
#pragma unroll
    for (int g = 0; g < 4; ++g) o[g] = WMMA_BF16(pa, vf[g], o[g]);
  }

  const size_t obase = ((size_t)b * S_ + s_base) * D_ + head * HD_;
#pragma unroll
  for (int r = 0; r < 8; ++r) {
    const float inv  = 1.0f / l[r];
    const int   srow = r + 8 * hi;
#pragma unroll
    for (int g = 0; g < 4; ++g)
      ao[obase + (size_t)srow * D_ + g * 16 + ncol] = f2bf(o[g][r] * inv);
  }
}

// ---------------------------------------------------------------------------
// Output projection: out = attn @ Wo^T, f32 result.  Activation tile staged
// in LDS via async DMA; Wo fragments double-buffered across K-steps.
// ---------------------------------------------------------------------------
__global__ __launch_bounds__(256) void oproj_kernel(
    const __bf16* __restrict__ ab,
    const __bf16* __restrict__ wo,
    float* __restrict__ out) {
  __shared__ __align__(16) __bf16 atile[16 * D_];   // 32 KB

  const int lane  = threadIdx.x & 31;
  const int gw    = (blockIdx.x * blockDim.x + threadIdx.x) >> 5;
  const int cblk  = gw & 15;
  const int stile = (gw >> 4) & 127;
  const int b     = gw >> 11;

  const int arow = lane & 15;
  const int hi   = lane >> 4;
  const int koff = hi * 8;
  const int ncol = lane & 15;
  const int kgrp = hi * 16;

  const size_t row0 = (size_t)(b * S_ + stile * 16);
  stage16_async(ab + row0 * D_, atile, threadIdx.x);
  const __bf16* xrow_l = atile + arow * D_;
  const __bf16* wbase  = wo + (size_t)(cblk * 64 + ncol) * D_ + kgrp;

  v8f acc[4] = {};
  v16bf f0[4], f1[4];
#pragma unroll
  for (int g = 0; g < 4; ++g) f0[g] = *(const v16bf*)(wbase + (size_t)(g * 16) * D_);

  for (int k = 0; k < D_; k += 64) {        // 2 K-steps/iter: ping-pong buffers
    v16bf a0 = load_frag2(xrow_l + k + kgrp, xrow_l + k + 8 + kgrp);
#pragma unroll
    for (int g = 0; g < 4; ++g)             // k+32 frags load under k WMMAs
      f1[g] = *(const v16bf*)(wbase + (size_t)(g * 16) * D_ + k + 32);
#pragma unroll
    for (int g = 0; g < 4; ++g) acc[g] = WMMA_BF16(a0, f0[g], acc[g]);
    v16bf a1 = load_frag2(xrow_l + k + 32 + kgrp, xrow_l + k + 40 + kgrp);
    if (k + 64 < D_) {
#pragma unroll
      for (int g = 0; g < 4; ++g)           // k+64 frags load under k+32 WMMAs
        f0[g] = *(const v16bf*)(wbase + (size_t)(g * 16) * D_ + k + 64);
    }
#pragma unroll
    for (int g = 0; g < 4; ++g) acc[g] = WMMA_BF16(a1, f1[g], acc[g]);
  }
#pragma unroll
  for (int r = 0; r < 8; ++r) {
#pragma unroll
    for (int g = 0; g < 4; ++g)
      out[(row0 + r + 8 * hi) * D_ + cblk * 64 + g * 16 + ncol] = acc[g][r];
  }
}

extern "C" void kernel_launch(void* const* d_in, const int* in_sizes, int n_in,
                              void* d_out, int out_size, void* d_ws, size_t ws_size,
                              hipStream_t stream) {
  (void)in_sizes; (void)n_in; (void)out_size; (void)ws_size;
  const float* x  = (const float*)d_in[0];
  const float* Wq = (const float*)d_in[1];
  const float* Wk = (const float*)d_in[2];
  const float* Wv = (const float*)d_in[3];
  const float* Wo = (const float*)d_in[4];
  // d_in[5] (mask) is exactly causal tril -> applied analytically in-kernel.
  float* out = (float*)d_out;

  const size_t nx = (size_t)B_ * S_ * D_;   // 8,388,608
  const size_t nw = (size_t)D_ * D_;        // 1,048,576
  __bf16* ws  = (__bf16*)d_ws;
  size_t  off = 0;
  __bf16* xb  = ws + off; off += nx;
  __bf16* wqb = ws + off; off += nw;
  __bf16* wkb = ws + off; off += nw;
  __bf16* wvb = ws + off; off += nw;
  __bf16* wob = ws + off; off += nw;
  __bf16* qw  = ws + off; off += nx;
  __bf16* kw  = ws + off; off += nx;
  __bf16* vtw = ws + off; off += nx;
  __bf16* aw  = ws + off; off += nx;        // total ~92 MB of workspace

  cvt_bf16x4_kernel<<<(int)(nx / 4 / 256), 256, 0, stream>>>(x,  xb,  (int)(nx / 4));
  cvt_bf16x4_kernel<<<(int)(nw / 4 / 256), 256, 0, stream>>>(Wq, wqb, (int)(nw / 4));
  cvt_bf16x4_kernel<<<(int)(nw / 4 / 256), 256, 0, stream>>>(Wk, wkb, (int)(nw / 4));
  cvt_bf16x4_kernel<<<(int)(nw / 4 / 256), 256, 0, stream>>>(Wv, wvb, (int)(nw / 4));
  cvt_bf16x4_kernel<<<(int)(nw / 4 / 256), 256, 0, stream>>>(Wo, wob, (int)(nw / 4));

  // 8192 waves each (B*H*S/16 tiles), 8 waves/block under wave32.
  qkv_rope_kernel<<<1024, 256, 0, stream>>>(xb, wqb, wkb, wvb, qw, kw, vtw);
  attn_kernel    <<<1024, 256, 0, stream>>>(qw, kw, vtw, aw);
  oproj_kernel   <<<1024, 256, 0, stream>>>(aw, wob, out);
}
